// OctreeResBlock_33973191311681
// MI455X (gfx1250) — matrix operand order repaired and checked by reading, not compiled, not run on hardware
//
#include <hip/hip_runtime.h>
#include <stdint.h>

typedef __attribute__((ext_vector_type(16))) __bf16 v16bf;
typedef __attribute__((ext_vector_type(8)))  float  v8f;
typedef int gvi4 __attribute__((vector_size(4 * sizeof(int))));   // builtin's vec type

#define C_IN  256
#define C_B   64
#define KNB   27
#define BN_EPS 1e-5f

#if __has_builtin(__builtin_amdgcn_global_load_async_to_lds_b128)
#define HAVE_ASYNC_LDS 1
#else
#define HAVE_ASYNC_LDS 0
#endif

// ---------- helpers ----------

__device__ __forceinline__ __bf16 f2bf(float f) {
    return (__bf16)f;                 // native RNE f32->bf16 convert
}

union BFrag {                 // one wave-wmma operand fragment, per-lane 16 bf16
    v16bf v;
    __bf16 e[16];
    uint4  q[2];
};

// B-operand per-lane K offsets (lane = column): lanes 0-15 K=0..7,16..23, lanes 16-31 K=8..15,24..31
__device__ __forceinline__ int bkoff(int lane, int j) {
    return (((lane >> 4) & 1) << 3) + j + (j >= 8 ? 8 : 0);
}

// ---------- weight prepack: f32 -> bf16 in WMMA-B fragment order ----------
// fragment f = B[kstep, coltile]; per lane 16 contiguous bf16 (32B)

__global__ void prepack_w1(const float* __restrict__ w1, __bf16* __restrict__ dst) {
    int f = blockIdx.x;            // 8 ksteps * 4 coltiles = 32 frags
    int lane = threadIdx.x;
    int kk = f >> 2, t = f & 3;
    __bf16* p = dst + (size_t)f * 512 + lane * 16;
    int col = t * 16 + (lane & 15);
#pragma unroll
    for (int j = 0; j < 16; ++j) {
        int c = kk * 32 + bkoff(lane, j);
        p[j] = f2bf(w1[c * C_B + col]);
    }
}

__global__ void prepack_w3(const float* __restrict__ w3, __bf16* __restrict__ dst) {
    int f = blockIdx.x;            // (27*2 ksteps) * 4 coltiles = 216 frags
    int lane = threadIdx.x;
    int t = f & 3, ks2 = f >> 2;   // ks2 = k*2 + ksub
    int ksub = ks2 & 1, k = ks2 >> 1;
    __bf16* p = dst + (size_t)f * 512 + lane * 16;
    int col = t * 16 + (lane & 15);
#pragma unroll
    for (int j = 0; j < 16; ++j) {
        int c = ksub * 32 + bkoff(lane, j);
        p[j] = f2bf(w3[k * (C_B * C_B) + c * C_B + col]);
    }
}

__global__ void prepack_w2(const float* __restrict__ w2, __bf16* __restrict__ dst) {
    int f = blockIdx.x;            // 2 ksteps * 16 coltiles = 32 frags
    int lane = threadIdx.x;
    int t = f & 15, ksub = f >> 4;
    __bf16* p = dst + (size_t)f * 512 + lane * 16;
    int col = t * 16 + (lane & 15);
#pragma unroll
    for (int j = 0; j < 16; ++j) {
        int c = ksub * 32 + bkoff(lane, j);
        p[j] = f2bf(w2[c * C_IN + col]);
    }
}

// ---------- K1: c1 = relu(bn1(data @ w1)) -> bf16 [N, 64] ----------
// block = 256 thr = 8 waves; each wave owns 16 rows and ALL 4 column tiles:
// A fragment loaded+converted once, reused by 4 wmma.

__global__ __launch_bounds__(256) void k1_conv1x1a(
    const float* __restrict__ data, const __bf16* __restrict__ w1p,
    const float* __restrict__ g, const float* __restrict__ b,
    const float* __restrict__ mu, const float* __restrict__ var,
    __bf16* __restrict__ c1bf)
{
    int lane = threadIdx.x & 31, wv = threadIdx.x >> 5;
    int baseRow = blockIdx.x * 128 + wv * 16;
    int mrow = baseRow + (lane & 15);
    int hi8 = ((lane >> 4) & 1) * 8;

    v8f acc[4] = {{}, {}, {}, {}};
#pragma unroll
    for (int kk = 0; kk < 8; ++kk) {
        int c0 = kk * 32 + hi8;
        const float4* p0 = (const float4*)(data + (size_t)mrow * C_IN + c0);
        const float4* p1 = (const float4*)(data + (size_t)mrow * C_IN + c0 + 16);
        float xs[16];
        *(float4*)&xs[0]  = p0[0]; *(float4*)&xs[4]  = p0[1];
        *(float4*)&xs[8]  = p1[0]; *(float4*)&xs[12] = p1[1];
        BFrag a;
#pragma unroll
        for (int j = 0; j < 16; ++j) a.e[j] = f2bf(xs[j]);
#pragma unroll
        for (int t = 0; t < 4; ++t) {
            BFrag bf;
            const uint4* bp = (const uint4*)(w1p + (size_t)(kk * 4 + t) * 512 + lane * 16);
            bf.q[0] = bp[0]; bf.q[1] = bp[1];
            acc[t] = __builtin_amdgcn_wmma_f32_16x16x32_bf16(
                false, a.v, false, bf.v, (short)0, acc[t], false, false);
        }
    }

    int mbase = baseRow + hi8;
#pragma unroll
    for (int t = 0; t < 4; ++t) {
        int col = t * 16 + (lane & 15);
        float sc = g[col] * rsqrtf(var[col] + BN_EPS);
        float sh = b[col] - mu[col] * sc;
#pragma unroll
        for (int r = 0; r < 8; ++r) {
            float val = acc[t][r] * sc + sh;
            val = val > 0.0f ? val : 0.0f;
            c1bf[(size_t)(mbase + r) * C_B + col] = f2bf(val);
        }
    }
}

// ---------- K2: gather 27 neighbors + einsum('nkc,kcd->nd') + bn2 + relu ----------
// block = 128 thr = 4 waves; 32 rows/block staged in dynamic LDS (padded pitch
// 72 bf16 -> conflict-free ds_load_b128). Each wave: 1 column tile x 2 row
// tiles, so each B fragment feeds 2 wmma.

#define K2_ROWS  32
#define K2_PITCH 72   // bf16 elems per gathered row (64 data + 8 pad)

__global__ __launch_bounds__(128) void k2_octree_conv(
    const __bf16* __restrict__ c1bf, const int* __restrict__ neigh,
    const __bf16* __restrict__ w3p,
    const float* __restrict__ g, const float* __restrict__ b,
    const float* __restrict__ mu, const float* __restrict__ var,
    __bf16* __restrict__ c2bf)
{
    extern __shared__ __align__(16) __bf16 smem[];   // 32*27*72*2 = 124416 B
    int tid = threadIdx.x;
    int lane = tid & 31, wv = tid >> 5;
    int blockBase = blockIdx.x * K2_ROWS;

    // cooperative gather: 864 rows x 128B payload; 8 threads/row, 16B each
    int sub = tid & 7;
    int rq  = tid >> 3;                       // 0..15
    for (int p = 0; p < (K2_ROWS * KNB) / 16; ++p) {   // 54 passes
        int r = p * 16 + rq;                  // 0..863 = m*27 + k
        int mm = r / KNB, kk = r - mm * KNB;
        int node = neigh[(size_t)(blockBase + mm) * KNB + kk];
        const uint4* src = (const uint4*)(c1bf + (size_t)node * C_B) + sub;
        uint4* dst = (uint4*)(smem + r * K2_PITCH) + sub;
#if HAVE_ASYNC_LDS
        __builtin_amdgcn_global_load_async_to_lds_b128(
            (gvi4*)src, (gvi4*)dst, 0, 0);
#else
        *dst = *src;
#endif
    }
#if HAVE_ASYNC_LDS
#if __has_builtin(__builtin_amdgcn_s_wait_asynccnt)
    __builtin_amdgcn_s_wait_asynccnt(0);
#else
    asm volatile("s_wait_asynccnt 0" ::: "memory");
#endif
#endif
    __syncthreads();

    int colTile = wv;
    int hi8 = ((lane >> 4) & 1) * 8;
    int m0  = lane & 15;
    v8f acc0 = {}, acc1 = {};
    for (int k = 0; k < KNB; ++k) {
#pragma unroll
        for (int ks = 0; ks < 2; ++ks) {
            const __bf16* a0p = smem + (m0 * KNB + k) * K2_PITCH + ks * 32 + hi8;
            const __bf16* a1p = a0p + 16 * KNB * K2_PITCH;
            BFrag a0, a1, bf;
            a0.q[0] = *(const uint4*)a0p;  a0.q[1] = *(const uint4*)(a0p + 16);
            a1.q[0] = *(const uint4*)a1p;  a1.q[1] = *(const uint4*)(a1p + 16);
            const uint4* bp = (const uint4*)(w3p +
                (size_t)((k * 2 + ks) * 4 + colTile) * 512 + lane * 16);
            bf.q[0] = bp[0]; bf.q[1] = bp[1];
            acc0 = __builtin_amdgcn_wmma_f32_16x16x32_bf16(
                false, a0.v, false, bf.v, (short)0, acc0, false, false);
            acc1 = __builtin_amdgcn_wmma_f32_16x16x32_bf16(
                false, a1.v, false, bf.v, (short)0, acc1, false, false);
        }
    }

    int col = colTile * 16 + (lane & 15);
    float sc = g[col] * rsqrtf(var[col] + BN_EPS);
    float sh = b[col] - mu[col] * sc;
    int mb0 = blockBase + hi8;
    int mb1 = blockBase + 16 + hi8;
#pragma unroll
    for (int r = 0; r < 8; ++r) {
        float v0 = acc0[r] * sc + sh;  v0 = v0 > 0.0f ? v0 : 0.0f;
        float v1 = acc1[r] * sc + sh;  v1 = v1 > 0.0f ? v1 : 0.0f;
        c2bf[(size_t)(mb0 + r) * C_B + col] = f2bf(v0);
        c2bf[(size_t)(mb1 + r) * C_B + col] = f2bf(v1);
    }
}

// ---------- K3: out = relu(bn3(c2 @ w2) + data) ----------
// block = 256 thr = 8 waves; 16 rows/block; each wave does coltiles wv, wv+8;
// A fragment reused for both tiles (2 wmma per k-step)

__global__ __launch_bounds__(256) void k3_conv1x1b(
    const __bf16* __restrict__ c2bf, const __bf16* __restrict__ w2p,
    const float* __restrict__ data,
    const float* __restrict__ g, const float* __restrict__ b,
    const float* __restrict__ mu, const float* __restrict__ var,
    float* __restrict__ out)
{
    int lane = threadIdx.x & 31, wv = threadIdx.x >> 5;
    int base = blockIdx.x * 16;
    int mrow = base + (lane & 15);
    int hi8 = ((lane >> 4) & 1) * 8;
    int t0 = wv, t1 = wv + 8;

    v8f acc0 = {}, acc1 = {};
#pragma unroll
    for (int ks = 0; ks < 2; ++ks) {
        const __bf16* ap = c2bf + (size_t)mrow * C_B + ks * 32 + hi8;
        BFrag a;
        a.q[0] = *(const uint4*)ap;
        a.q[1] = *(const uint4*)(ap + 16);
        BFrag b0, b1;
        const uint4* bp0 = (const uint4*)(w2p + (size_t)(ks * 16 + t0) * 512 + lane * 16);
        const uint4* bp1 = (const uint4*)(w2p + (size_t)(ks * 16 + t1) * 512 + lane * 16);
        b0.q[0] = bp0[0]; b0.q[1] = bp0[1];
        b1.q[0] = bp1[0]; b1.q[1] = bp1[1];
        acc0 = __builtin_amdgcn_wmma_f32_16x16x32_bf16(
            false, a.v, false, b0.v, (short)0, acc0, false, false);
        acc1 = __builtin_amdgcn_wmma_f32_16x16x32_bf16(
            false, a.v, false, b1.v, (short)0, acc1, false, false);
    }

    int mbase = base + hi8;
#pragma unroll
    for (int tt = 0; tt < 2; ++tt) {
        int t = tt ? t1 : t0;
        v8f acc = tt ? acc1 : acc0;
        int col = t * 16 + (lane & 15);
        float sc = g[col] * rsqrtf(var[col] + BN_EPS);
        float sh = b[col] - mu[col] * sc;
#pragma unroll
        for (int r = 0; r < 8; ++r) {
            float val = acc[r] * sc + sh + data[(size_t)(mbase + r) * C_IN + col];
            val = val > 0.0f ? val : 0.0f;
            out[(size_t)(mbase + r) * C_IN + col] = val;
        }
    }
}

// ---------- launcher ----------

extern "C" void kernel_launch(void* const* d_in, const int* in_sizes, int n_in,
                              void* d_out, int out_size, void* d_ws, size_t ws_size,
                              hipStream_t stream) {
    const float* data = (const float*)d_in[0];
    const int*   neigh = (const int*)d_in[1];
    // d_in[2] = depth (unused)
    const float* w1   = (const float*)d_in[3];
    const float* bn1g = (const float*)d_in[4];
    const float* bn1b = (const float*)d_in[5];
    const float* bn1m = (const float*)d_in[6];
    const float* bn1v = (const float*)d_in[7];
    const float* w3   = (const float*)d_in[8];
    const float* bn2g = (const float*)d_in[9];
    const float* bn2b = (const float*)d_in[10];
    const float* bn2m = (const float*)d_in[11];
    const float* bn2v = (const float*)d_in[12];
    const float* w2   = (const float*)d_in[13];
    const float* bn3g = (const float*)d_in[14];
    const float* bn3b = (const float*)d_in[15];
    const float* bn3m = (const float*)d_in[16];
    const float* bn3v = (const float*)d_in[17];
    float* out = (float*)d_out;

    const int N = in_sizes[0] / C_IN;    // 131072

    char* ws = (char*)d_ws;
    __bf16* w1p  = (__bf16*)(ws);                         //  32 KB
    __bf16* w2p  = (__bf16*)(ws + (32u << 10));           //  32 KB
    __bf16* w3p  = (__bf16*)(ws + (64u << 10));           // 216 KB
    __bf16* c1bf = (__bf16*)(ws + (512u << 10));          // N*64 bf16 = 16 MB
    __bf16* c2bf = (__bf16*)(ws + (512u << 10) + (size_t)N * C_B * sizeof(__bf16));

    prepack_w1<<<32, 32, 0, stream>>>(w1, w1p);
    prepack_w2<<<32, 32, 0, stream>>>(w2, w2p);
    prepack_w3<<<KNB * 2 * 4, 32, 0, stream>>>(w3, w3p);

    k1_conv1x1a<<<N / 128, 256, 0, stream>>>(data, w1p, bn1g, bn1b, bn1m, bn1v, c1bf);

    const int k2_lds = K2_ROWS * KNB * K2_PITCH * 2;      // 124416 B (<320KB/WGP)
    k2_octree_conv<<<N / K2_ROWS, 128, k2_lds, stream>>>(c1bf, neigh, w3p,
                                                         bn2g, bn2b, bn2m, bn2v, c2bf);

    k3_conv1x1b<<<N / 16, 256, 0, stream>>>(c2bf, w2p, data,
                                            bn3g, bn3b, bn3m, bn3v, out);
}
